// Discrimitor_37873021616660
// MI455X (gfx1250) — matrix operand order repaired and checked by reading, not compiled, not run on hardware
//
#include <hip/hip_runtime.h>

typedef __attribute__((ext_vector_type(16))) _Float16 v16h;
typedef __attribute__((ext_vector_type(8)))  _Float16 v8h;
typedef __attribute__((ext_vector_type(4)))  _Float16 v4h;
typedef __attribute__((ext_vector_type(8)))  float    v8f;

#define EMBED   100
#define KDIM    300           // 3 * EMBED
#define KPAD    320           // padded to multiple of 32 for WMMA K-steps
#define LSTRIDE 328           // padded row stride in halfs (656B = 41*16, bank-spread)
#define HHALF   64            // hidden width (W1 columns)
#define WAVES   8             // waves per workgroup
#define BATCH   524288
#define NTILES  (BATCH / 16)
#define NQ      (EMBED / 4)   // 25 float4 per embedding row

__global__ __launch_bounds__(256) void disc_fused_kernel(
    const int*   __restrict__ anchor,
    const int*   __restrict__ cand,
    const float* __restrict__ doc,
    const float* __restrict__ W1,
    const float* __restrict__ b1,
    const float* __restrict__ W2,
    const float* __restrict__ b2,
    float*       __restrict__ out)
{
    // W1 transposed to [64][LSTRIDE] f16 (column-contiguous in K) + per-wave hidden tiles
    __shared__ _Float16 sW1[HHALF * LSTRIDE];
    __shared__ _Float16 sHid[WAVES * 16 * LSTRIDE];

    const int tid  = threadIdx.x;
    const int lane = tid & 31;
    const int wave = tid >> 5;
    const int nlo  = lane & 15;          // N within a 16-wide tile / M-row selector
    const int khi  = lane >> 4;          // half-wave flag (K-split in A/B fragments)

    // ---- Stage W1 into LDS: transposed, f16, K padded with zeros (one-time) ----
    for (int i = tid; i < HHALF * (LSTRIDE - KDIM); i += 256) {
        int n = i / (LSTRIDE - KDIM);
        int k = KDIM + i % (LSTRIDE - KDIM);
        sW1[n * LSTRIDE + k] = (_Float16)0.0f;
    }
    // fill: W1 global is row-major [300][64]
    for (int i = tid; i < KDIM * HHALF; i += 256) {
        int k = i / HHALF;
        int n = i % HHALF;
        sW1[n * LSTRIDE + k] = (_Float16)W1[i];
    }
    __syncthreads();

    // ---- Per-wave hidden tile; zero the K pad columns once (they never change) ----
    _Float16* hid = &sHid[wave * 16 * LSTRIDE];
    for (int i = lane; i < 16 * (LSTRIDE - KDIM); i += 32) {
        int r = i / (LSTRIDE - KDIM);
        int c = KDIM + i % (LSTRIDE - KDIM);
        hid[r * LSTRIDE + c] = (_Float16)0.0f;
    }

    // ---- Per-lane epilogue constants: N = 16*t + nlo is tile-invariant ----
    float b1v[4], w2v[4];
#pragma unroll
    for (int t = 0; t < 4; ++t) {
        b1v[t] = b1[t * 16 + nlo];
        w2v[t] = W2[t * 16 + nlo];
    }
    const float b2v = b2[0];

    const int waveGlobal = blockIdx.x * WAVES + wave;
    const int waveCount  = gridDim.x * WAVES;
    const int q = lane;                  // float4 slot within an embedding row (q < 25)

    for (int tile = waveGlobal; tile < NTILES; tile += waveCount) {
        const int rowBase = tile * 16;

        // lanes 0-15 hold anchor indices, lanes 16-31 candidate indices
        int myIdx = (lane < 16) ? anchor[rowBase + lane]
                                : cand[rowBase + (lane - 16)];

        // ---- Gather 16 rows (float4 wide), build hidden = [a | c | a*c] in f16 LDS ----
        // 4-row batches: 8 independent global_load_b128 in flight before conversion.
        for (int r0 = 0; r0 < 16; r0 += 4) {
            float4 va[4], vc[4];
#pragma unroll
            for (int r = 0; r < 4; ++r) {
                int ia = __shfl(myIdx, r0 + r, 32);
                int ic = __shfl(myIdx, 16 + r0 + r, 32);
                if (q < NQ) {
                    va[r] = ((const float4*)(doc + (size_t)ia * EMBED))[q];
                    vc[r] = ((const float4*)(doc + (size_t)ic * EMBED))[q];
                }
            }
#pragma unroll
            for (int r = 0; r < 4; ++r) {
                if (q < NQ) {
                    _Float16* hrow = hid + (r0 + r) * LSTRIDE;
                    v4h ha, hc, hp;
                    ha[0] = (_Float16)va[r].x; ha[1] = (_Float16)va[r].y;
                    ha[2] = (_Float16)va[r].z; ha[3] = (_Float16)va[r].w;
                    hc[0] = (_Float16)vc[r].x; hc[1] = (_Float16)vc[r].y;
                    hc[2] = (_Float16)vc[r].z; hc[3] = (_Float16)vc[r].w;
                    hp[0] = (_Float16)(va[r].x * vc[r].x);
                    hp[1] = (_Float16)(va[r].y * vc[r].y);
                    hp[2] = (_Float16)(va[r].z * vc[r].z);
                    hp[3] = (_Float16)(va[r].w * vc[r].w);
                    *(v4h*)(hrow + 4 * q)             = ha;
                    *(v4h*)(hrow + EMBED + 4 * q)     = hc;
                    *(v4h*)(hrow + 2 * EMBED + 4 * q) = hp;
                }
            }
        }
        // LDS ops from the same wave execute in order: RAW through LDS is safe.

        // ---- GEMM: [16 x 320] x [320 x 64] via v_wmma_f32_16x16x32_f16 ----
        v8f zero = {0.f, 0.f, 0.f, 0.f, 0.f, 0.f, 0.f, 0.f};
        v8f acc[4];
#pragma unroll
        for (int t = 0; t < 4; ++t) acc[t] = zero;

        // unroll capped at 2: keeps ~1 A + 8 B fragments live -> lower VGPR
        // pressure -> higher occupancy for the gather-latency-bound phase.
#pragma unroll 2
        for (int k = 0; k < KPAD / 32; ++k) {
            // A fragment: lane -> row (lane&15); halves 0..7 at K=8*khi, 8..15 at +16
            const _Float16* arow = hid + nlo * LSTRIDE + k * 32 + khi * 8;
            v8h alo = *(const v8h*)(arow);
            v8h ahi = *(const v8h*)(arow + 16);
            v16h a;
#pragma unroll
            for (int i = 0; i < 8; ++i) { a[i] = alo[i]; a[8 + i] = ahi[i]; }

#pragma unroll
            for (int t = 0; t < 4; ++t) {
                // B fragment: lane -> col N = 16t + nlo; halves 0..7 at K=8*khi, 8..15 at +16
                const _Float16* bcol = sW1 + (t * 16 + nlo) * LSTRIDE + k * 32 + khi * 8;
                v8h blo = *(const v8h*)(bcol);
                v8h bhi = *(const v8h*)(bcol + 16);
                v16h bfrag;
#pragma unroll
                for (int i = 0; i < 8; ++i) { bfrag[i] = blo[i]; bfrag[8 + i] = bhi[i]; }

                acc[t] = __builtin_amdgcn_wmma_f32_16x16x32_f16(
                    /*neg_a=*/false, a, /*neg_b=*/false, bfrag,
                    /*c_mod=*/(short)0, acc[t],
                    /*reuse_a=*/false, /*reuse_b=*/false);
            }
        }

        // ---- Epilogue: +b1, ReLU, *W2, reduce over N (16 lanes), +b2, store ----
        // C layout: VGPR r, lanes 0-15 -> M=r, N=lane; lanes 16-31 -> M=r+8, N=lane-16
        float psum[8];
#pragma unroll
        for (int r = 0; r < 8; ++r) psum[r] = 0.0f;
#pragma unroll
        for (int t = 0; t < 4; ++t) {
#pragma unroll
            for (int r = 0; r < 8; ++r) {
                float h1 = acc[t][r] + b1v[t];
                h1 = h1 > 0.0f ? h1 : 0.0f;
                psum[r] += h1 * w2v[t];
            }
        }
#pragma unroll
        for (int r = 0; r < 8; ++r) {
            float s = psum[r];
            s += __shfl_xor(s, 1, 32);
            s += __shfl_xor(s, 2, 32);
            s += __shfl_xor(s, 4, 32);
            s += __shfl_xor(s, 8, 32);   // reduction stays within each 16-lane half
            if (nlo == 0) {
                out[rowBase + r + khi * 8] = s + b2v;
            }
        }
    }
}

extern "C" void kernel_launch(void* const* d_in, const int* in_sizes, int n_in,
                              void* d_out, int out_size, void* d_ws, size_t ws_size,
                              hipStream_t stream) {
    const int*   anchor = (const int*)  d_in[0];
    const int*   cand   = (const int*)  d_in[1];
    const float* doc    = (const float*)d_in[2];
    const float* W1     = (const float*)d_in[3];
    const float* b1     = (const float*)d_in[4];
    const float* W2     = (const float*)d_in[5];
    const float* b2     = (const float*)d_in[6];
    float* out = (float*)d_out;

    // 2048 blocks x 8 waves = 16384 waves; 32768 tiles -> exactly 2 tiles/wave
    disc_fused_kernel<<<dim3(2048), dim3(256), 0, stream>>>(
        anchor, cand, doc, W1, b1, W2, b2, out);
}
